// SpatialTemporalAttention_59313498358146
// MI455X (gfx1250) — matrix operand order repaired and checked by reading, not compiled, not run on hardware
//
#include <hip/hip_runtime.h>

// ---------------------------------------------------------------------------
// SpatialTemporalAttention for MI455X (gfx1250): full bf16-WMMA pipeline.
// B=2, S=2048, D=1024, H=16, Dh=64. fp32 accumulate everywhere.
// ---------------------------------------------------------------------------

#define SEQ      2048
#define DMODEL   1024
#define NHEADS   16
#define HEADDIM  64
#define NTOK     (2 * SEQ)          // B*S = 4096
#define PERBH    ((size_t)SEQ * HEADDIM)  // 131072 elements per (b,h) pack

typedef __attribute__((ext_vector_type(16))) __bf16 v16bf;
typedef __attribute__((ext_vector_type(8)))  __bf16 bf16x8;
typedef __attribute__((ext_vector_type(8)))  float  v8f;

static __device__ inline v8f wmma_bf16(v16bf a, v16bf b, v8f c) {
  // D = A(16x32 bf16) * B(32x16 bf16) + C(16x16 f32)
  return __builtin_amdgcn_wmma_f32_16x16x32_bf16(false, a, false, b,
                                                 (short)0, c, false, false);
}

static __device__ inline v16bf make16(bf16x8 lo, bf16x8 hi) {
  v16bf r;
#pragma unroll
  for (int i = 0; i < 8; ++i) { r[i] = lo[i]; r[i + 8] = hi[i]; }
  return r;
}

// A-fragment (16x32 bf16) from a row-major [*, ldm] bf16 matrix.
// lane<16: row=row0+lane, k = {col..col+15}
// lane>=16: row=row0+lane-16, k = {col+16..col+31}
static __device__ inline v16bf load_a_frag(const __bf16* base, int ldm,
                                           int row0, int col, int lane) {
  int r = row0 + (lane & 15);
  int c = col + ((lane >> 4) << 4);
  const __bf16* p = base + (size_t)r * ldm + c;
  bf16x8 lo = *(const bf16x8*)p;
  bf16x8 hi = *(const bf16x8*)(p + 8);
  return make16(lo, hi);
}

// B-fragment (32x16 bf16) from pre-packed storage: element lane*16+j.
static __device__ inline v16bf load_b_frag(const __bf16* fragBase, int lane) {
  const __bf16* p = fragBase + lane * 16;
  bf16x8 lo = *(const bf16x8*)p;
  bf16x8 hi = *(const bf16x8*)(p + 8);
  return make16(lo, hi);
}

// ---------------------------------------------------------------------------
// Stage 0: fp32 -> bf16 copy
// ---------------------------------------------------------------------------
__global__ void to_bf16_kernel(const float* __restrict__ src,
                               __bf16* __restrict__ dst, int n) {
  int i = blockIdx.x * blockDim.x + threadIdx.x;
  if (i < n) dst[i] = (__bf16)src[i];
}

// ---------------------------------------------------------------------------
// Stage 0b: pack a [K=1024, N=1024] fp32 weight into B-fragment layout:
//   flat = (nTile*32 + kChunk)*512 + lane*16 + j
//   k = kChunk*32 + 16*(lane/16) + j ; n = nTile*16 + (lane%16)
// ---------------------------------------------------------------------------
__global__ void pack_w_kernel(const float* __restrict__ W,
                              __bf16* __restrict__ Wp) {
  int e = blockIdx.x * blockDim.x + threadIdx.x;   // 0 .. 1M-1
  int j      = e & 15;
  int lp     = (e >> 4) & 31;
  int kChunk = (e >> 9) & 31;
  int nTile  = e >> 14;
  int k = kChunk * 32 + ((lp >> 4) << 4) + j;
  int n = nTile * 16 + (lp & 15);
  Wp[e] = (__bf16)W[(size_t)k * DMODEL + n];
}

// ---------------------------------------------------------------------------
// Tiled WMMA GEMM: C[M=4096, N=1024] = A[4096,1024] * Bpack
// 256 threads = 8 waves; wave tile 64x64 (4x4 of 16x16); block tile 128x256.
// Epilogue modes: 0=Q row-major bf16, 1=K scatter to QK^T B-frag layout,
//                 2=V scatter to PV B-frag layout, 3=f32 out + bias (Wo).
// ---------------------------------------------------------------------------
__global__ __launch_bounds__(256) void gemm_wmma_kernel(
    const __bf16* __restrict__ A, const __bf16* __restrict__ Bp,
    __bf16* __restrict__ outB, float* __restrict__ outF,
    const float* __restrict__ bias, int mode) {
  const int lane = threadIdx.x & 31;
  const int wid  = threadIdx.x >> 5;
  const int wm = wid >> 2, wn = wid & 3;
  const int m0 = blockIdx.x * 128 + wm * 64;
  const int n0 = blockIdx.y * 256 + wn * 64;

  const v8f vz = {0.f, 0.f, 0.f, 0.f, 0.f, 0.f, 0.f, 0.f};
  v8f acc[16];
#pragma unroll
  for (int i = 0; i < 16; ++i) acc[i] = vz;

  for (int kc = 0; kc < 32; ++kc) {
    v16bf af[4];
#pragma unroll
    for (int mi = 0; mi < 4; ++mi)
      af[mi] = load_a_frag(A, DMODEL, m0 + mi * 16, kc * 32, lane);
#pragma unroll
    for (int ni = 0; ni < 4; ++ni) {
      int nTile = (n0 + ni * 16) >> 4;
      v16bf bf = load_b_frag(Bp + ((size_t)nTile * 32 + kc) * 512, lane);
#pragma unroll
      for (int mi = 0; mi < 4; ++mi)
        acc[mi * 4 + ni] = wmma_bf16(af[mi], bf, acc[mi * 4 + ni]);
    }
  }

#pragma unroll
  for (int mi = 0; mi < 4; ++mi) {
#pragma unroll
    for (int ni = 0; ni < 4; ++ni) {
      v8f c = acc[mi * 4 + ni];
#pragma unroll
      for (int v = 0; v < 8; ++v) {
        int row = m0 + mi * 16 + v + ((lane >> 4) << 3);
        int n   = n0 + ni * 16 + (lane & 15);
        float val = c[v];
        if (mode == 0) {
          outB[(size_t)row * DMODEL + n] = (__bf16)val;
        } else if (mode == 1) {
          // K pack: per (b,h): [sTile(128)][dChunk(2)] frags of 512.
          int b = row >> 11, s = row & 2047;
          int h = n >> 6,  d = n & 63;
          int dChunk = d >> 5, dsub = d & 31;
          int lp = (s & 15) + ((dsub >> 4) << 4);
          int j  = dsub & 15;
          size_t idx = (size_t)(b * NHEADS + h) * PERBH +
                       ((size_t)((s >> 4) * 2 + dChunk)) * 512 + lp * 16 + j;
          outB[idx] = (__bf16)val;
        } else if (mode == 2) {
          // V pack: per (b,h): [sChunk(64)][dTile(4)] frags of 512.
          int b = row >> 11, s = row & 2047;
          int h = n >> 6,  d = n & 63;
          int sChunk = s >> 5, ssub = s & 31;
          int lp = (d & 15) + ((ssub >> 4) << 4);
          int j  = ssub & 15;
          size_t idx = (size_t)(b * NHEADS + h) * PERBH +
                       ((size_t)(sChunk * 4 + (d >> 4))) * 512 + lp * 16 + j;
          outB[idx] = (__bf16)val;
        } else {
          outF[(size_t)row * DMODEL + n] = val + bias[n];
        }
      }
    }
  }
}

// ---------------------------------------------------------------------------
// Flash-attention: one wave per (b, h, 16-row q-tile). Online softmax,
// spatial bias sb*exp(-dist), mask, P restriped through per-wave LDS tile.
// ---------------------------------------------------------------------------
__global__ __launch_bounds__(256) void attn_wmma_kernel(
    const __bf16* __restrict__ Qb, const __bf16* __restrict__ Kp,
    const __bf16* __restrict__ Vp, const float* __restrict__ coords,
    const int* __restrict__ mask, const float* __restrict__ sbias,
    __bf16* __restrict__ attOut) {
  __shared__ __align__(16) __bf16 lds[8][16 * 32];

  const int lane = threadIdx.x & 31;
  const int wid  = threadIdx.x >> 5;
  const int gw = blockIdx.x * 8 + wid;          // 0..4095
  const int qt = gw & 127;
  const int h  = (gw >> 7) & 15;
  const int b  = gw >> 11;
  const int q0 = qt * 16;
  const int halfOff = (lane >> 4) << 3;         // row offset of C-layout

  // Q fragments (16 rows x 64 head-dim = two 16x32 A-frags), row-major Qb.
  v16bf qf[2];
#pragma unroll
  for (int kc = 0; kc < 2; ++kc)
    qf[kc] = load_a_frag(Qb, DMODEL, b * SEQ + q0, h * HEADDIM + kc * 32, lane);

  // Query coords for this lane's 8 rows.
  float qcx[8], qcy[8];
#pragma unroll
  for (int v = 0; v < 8; ++v) {
    int r = q0 + v + halfOff;
    qcx[v] = coords[((size_t)b * SEQ + r) * 2 + 0];
    qcy[v] = coords[((size_t)b * SEQ + r) * 2 + 1];
  }
  const float sb = sbias[h];

  float m_[8], l_[8];
#pragma unroll
  for (int v = 0; v < 8; ++v) { m_[v] = -__builtin_inff(); l_[v] = 0.f; }
  const v8f vz = {0.f, 0.f, 0.f, 0.f, 0.f, 0.f, 0.f, 0.f};
  v8f accO[4];
#pragma unroll
  for (int dt = 0; dt < 4; ++dt) accO[dt] = vz;

  const __bf16* kBase = Kp + (size_t)(b * NHEADS + h) * PERBH;
  const __bf16* vBase = Vp + (size_t)(b * NHEADS + h) * PERBH;
  __bf16* myLds = &lds[wid][0];

  for (int jc = 0; jc < SEQ / 32; ++jc) {       // 32 keys per step
    float pvals[2][8];
    float tmax[8];
#pragma unroll
    for (int v = 0; v < 8; ++v) tmax[v] = -__builtin_inff();

#pragma unroll
    for (int t = 0; t < 2; ++t) {
      int sTile = jc * 2 + t;
      v16bf kf0 = load_b_frag(kBase + ((size_t)sTile * 2 + 0) * 512, lane);
      v16bf kf1 = load_b_frag(kBase + ((size_t)sTile * 2 + 1) * 512, lane);
      v8f sc = vz;
      sc = wmma_bf16(qf[0], kf0, sc);
      sc = wmma_bf16(qf[1], kf1, sc);

      int skey = jc * 32 + t * 16 + (lane & 15);
      float kx = coords[((size_t)b * SEQ + skey) * 2 + 0];
      float ky = coords[((size_t)b * SEQ + skey) * 2 + 1];
#pragma unroll
      for (int v = 0; v < 8; ++v) {
        float s = sc[v] * 0.125f;               // 1/sqrt(64)
        float dx = qcx[v] - kx, dy = qcy[v] - ky;
        float d2 = dx * dx + dy * dy;
        float dist = (d2 == 0.f) ? 0.f : sqrtf(d2);
        s += sb * __expf(-dist);
        int r = q0 + v + halfOff;
        int mv = mask[((size_t)b * SEQ + r) * SEQ + skey];
        if (mv == 0) s = -__builtin_inff();
        pvals[t][v] = s;
        tmax[v] = fmaxf(tmax[v], s);
      }
    }

    // Online softmax update (row stats live in 16-lane half groups).
#pragma unroll
    for (int v = 0; v < 8; ++v) {
      float mv = tmax[v];
#pragma unroll
      for (int off = 1; off < 16; off <<= 1)
        mv = fmaxf(mv, __shfl_xor(mv, off, 32));
      float mnew = fmaxf(m_[v], mv);
      float alpha = (m_[v] == -__builtin_inff()) ? 0.f : __expf(m_[v] - mnew);
      float rsum = 0.f;
#pragma unroll
      for (int t = 0; t < 2; ++t) {
        float p = __expf(pvals[t][v] - mnew);
        pvals[t][v] = p;
        rsum += p;
      }
#pragma unroll
      for (int off = 1; off < 16; off <<= 1)
        rsum += __shfl_xor(rsum, off, 32);
      l_[v] = l_[v] * alpha + rsum;
      m_[v] = mnew;
#pragma unroll
      for (int dt = 0; dt < 4; ++dt) accO[dt][v] *= alpha;
      // stage P (C-layout) into LDS 16x32 tile
      int r = v + halfOff;
      myLds[r * 32 + (lane & 15)]      = (__bf16)pvals[0][v];
      myLds[r * 32 + 16 + (lane & 15)] = (__bf16)pvals[1][v];
    }

    asm volatile("s_wait_dscnt 0" ::: "memory");

    // Re-stripe P from LDS into A-fragment layout.
    v16bf pf;
    {
      int r = lane & 15;
      int cbase = (lane >> 4) << 4;
      bf16x8 lo = *(const bf16x8*)&myLds[r * 32 + cbase];
      bf16x8 hi = *(const bf16x8*)&myLds[r * 32 + cbase + 8];
      pf = make16(lo, hi);
    }
#pragma unroll
    for (int dt = 0; dt < 4; ++dt) {
      v16bf vf = load_b_frag(vBase + ((size_t)jc * 4 + dt) * 512, lane);
      accO[dt] = wmma_bf16(pf, vf, accO[dt]);
    }
  }

  // Normalize and write attended [B,S,1024] bf16 (feeds Wo GEMM).
#pragma unroll
  for (int dt = 0; dt < 4; ++dt) {
#pragma unroll
    for (int v = 0; v < 8; ++v) {
      int r = q0 + v + halfOff;
      int d = h * HEADDIM + dt * 16 + (lane & 15);
      float o = accO[dt][v] / l_[v];
      attOut[((size_t)b * SEQ + r) * DMODEL + d] = (__bf16)o;
    }
  }
}

// ---------------------------------------------------------------------------
// Host-side orchestration. Workspace layout (48 MB total):
//   [0,8)MB xb | [8,16)MB Wq/Wk/Wv/Wo packs | [16,24) Qb | [24,32) Kpack
//   [32,40) Vpack | [40,48) attended
// ---------------------------------------------------------------------------
extern "C" void kernel_launch(void* const* d_in, const int* in_sizes, int n_in,
                              void* d_out, int out_size, void* d_ws,
                              size_t ws_size, hipStream_t stream) {
  (void)in_sizes; (void)n_in; (void)out_size; (void)ws_size;
  const float* x      = (const float*)d_in[0];
  const float* coords = (const float*)d_in[1];
  const int*   mask   = (const int*)d_in[2];
  const float* Wq     = (const float*)d_in[3];
  const float* Wk     = (const float*)d_in[4];
  const float* Wv     = (const float*)d_in[5];
  const float* sbias  = (const float*)d_in[6];
  const float* Wo     = (const float*)d_in[7];
  const float* bo     = (const float*)d_in[8];
  float* out = (float*)d_out;

  char* ws = (char*)d_ws;
  const size_t MB = 1u << 20;
  __bf16* xb  = (__bf16*)(ws + 0 * MB);
  __bf16* WqP = (__bf16*)(ws + 8 * MB);
  __bf16* WkP = (__bf16*)(ws + 10 * MB);
  __bf16* WvP = (__bf16*)(ws + 12 * MB);
  __bf16* WoP = (__bf16*)(ws + 14 * MB);
  __bf16* Qb  = (__bf16*)(ws + 16 * MB);
  __bf16* Kp  = (__bf16*)(ws + 24 * MB);
  __bf16* Vp  = (__bf16*)(ws + 32 * MB);
  __bf16* att = (__bf16*)(ws + 40 * MB);

  // Stage 0: conversions / weight packing.
  const int nX = NTOK * DMODEL;  // 4M
  to_bf16_kernel<<<(nX + 255) / 256, 256, 0, stream>>>(x, xb, nX);
  const int nW = DMODEL * DMODEL;  // 1M
  pack_w_kernel<<<nW / 256, 256, 0, stream>>>(Wq, WqP);
  pack_w_kernel<<<nW / 256, 256, 0, stream>>>(Wk, WkP);
  pack_w_kernel<<<nW / 256, 256, 0, stream>>>(Wv, WvP);
  pack_w_kernel<<<nW / 256, 256, 0, stream>>>(Wo, WoP);

  // Stage 1: QKV projections (K/V written straight into attention B-frag
  // layouts from the epilogue).
  dim3 gg(NTOK / 128, DMODEL / 256);
  gemm_wmma_kernel<<<gg, 256, 0, stream>>>(xb, WqP, Qb, nullptr, nullptr, 0);
  gemm_wmma_kernel<<<gg, 256, 0, stream>>>(xb, WkP, Kp, nullptr, nullptr, 1);
  gemm_wmma_kernel<<<gg, 256, 0, stream>>>(xb, WvP, Vp, nullptr, nullptr, 2);

  // Stage 2: fused flash attention. 4096 wave-tasks / 8 waves per block.
  attn_wmma_kernel<<<512, 256, 0, stream>>>(Qb, Kp, Vp, coords, mask, sbias,
                                            att);

  // Stage 3: output projection + bias, fp32 out.
  gemm_wmma_kernel<<<gg, 256, 0, stream>>>(att, WoP, nullptr, out, bo, 3);
}
